// SaliencyDinatBlock_68186900791920
// MI455X (gfx1250) — compile-verified, matrix operand order checked by reading, not run on hardware
//
#include <hip/hip_runtime.h>
#include <hip/hip_bf16.h>
#include <math.h>

// ---------------------------------------------------------------------------
// SaliencyDinatBlock for MI455X (gfx1250, wave32, WMMA + async-LDS staging)
//   B=2, H=W=56, DIM=128, HEADS=4 (dh=32), K=11, DIL=2, MLP_HID=512
// GEMMs: v_wmma_f32_16x16x32_f16; weight tile staged to LDS once per block
// via GLOBAL_LOAD_ASYNC_TO_LDS_B128 (ASYNCcnt), B fragments via ds_load.
// Attention: one wave32 per (pixel, head), lane == channel.
// ---------------------------------------------------------------------------

typedef __attribute__((ext_vector_type(16))) _Float16 v16h;
typedef __attribute__((ext_vector_type(8)))  float    v8f;

#define BB    2
#define HH    56
#define WWID  56
#define DIM   128
#define HEADS 4
#define DH    32
#define KK    11
#define DIL   2
#define MLP   512
#define NPIX  (BB * HH * WWID)   // 6272

// ---------------- wave32 helpers ----------------
__device__ __forceinline__ float wave_reduce_add(float v) {
#pragma unroll
  for (int off = 16; off > 0; off >>= 1) v += __shfl_xor(v, off, 32);
  return v;
}

// ---------------- WMMA fragment load ----------------
// 16-bit A (16x32) fragment, wave32 (ISA 7.12.2):
//   lane L: row = L&15, K-half = L>>4 (adds 8)
//   elements 0..7  -> K = khalf*8 + 0..7   (contiguous 16 bytes)
//   elements 8..15 -> K = khalf*8 + 16..23 (contiguous 16 bytes)
// B (32x16, K-major: W stored NxK row-major) has the identical per-lane
// layout with "row" = output column N, so one loader serves both operands.
__device__ __forceinline__ v16h load_frag(const _Float16* __restrict__ base) {
  v16h f;
#pragma unroll
  for (int e = 0; e < 8; ++e) f[e] = base[e];
#pragma unroll
  for (int e = 0; e < 8; ++e) f[8 + e] = base[16 + e];
  return f;
}

// ---------------- WMMA GEMM: out = A(MxKD) @ Wt(NxKD)^T (+bias)(+epilogue) --
// EPI 0: out f32 = acc (+bias)
// EPI 1: out f16 = gelu_exact(acc + bias)
// EPI 2: out f32 = acc + bias + res[m*N+n]
// One wave computes a 16(M) x 64(N) tile; block = 8 waves = 128 M-rows.
// The 64xKD weight tile (64 contiguous rows of Wt) is staged into LDS once
// per block with async global->LDS copies.
template <int EPI, int KD>
__global__ __launch_bounds__(256) void wmma_gemm(
    const _Float16* __restrict__ A, const _Float16* __restrict__ Wt,
    const float* __restrict__ bias, const float* __restrict__ res,
    void* __restrict__ outp, int M, int N) {
  __shared__ __align__(16) _Float16 bsh[64 * KD];

  const int lane  = threadIdx.x & 31;
  const int wave  = threadIdx.x >> 5;
  const int n0    = blockIdx.x * 64;
  const int m0    = (blockIdx.y * 8 + wave) * 16;
  const int row   = lane & 15;
  const int khalf = lane >> 4;

  // ---- async stage: 64 rows x KD f16 = contiguous 64*KD*2 bytes ----
  {
    const _Float16* gsrc = Wt + (size_t)n0 * KD;          // tile is contiguous
    const unsigned ldsbase = (unsigned)(uintptr_t)&bsh[0];
#pragma unroll
    for (int it = 0; it < KD / 32; ++it) {                // 256 thr * 16B each
      const int ci = it * 256 + threadIdx.x;              // 16-byte chunk id
      const unsigned loff = ldsbase + ci * 16;
      const _Float16* g = gsrc + ci * 8;
      asm volatile("global_load_async_to_lds_b128 %0, %1, off"
                   :: "v"(loff), "v"(g)
                   : "memory");
    }
    asm volatile("s_wait_asynccnt 0" ::: "memory");       // this wave's copies
    __syncthreads();                                      // cross-wave visibility
  }

  const _Float16* aRow = A + (size_t)(m0 + row) * KD + khalf * 8;
  const _Float16* bRow = &bsh[(size_t)row * KD + khalf * 8];  // LDS, generic

  v8f acc[4] = {};
#pragma unroll
  for (int k0 = 0; k0 < KD; k0 += 32) {
    v16h a  = load_frag(aRow + k0);
    v16h b0 = load_frag(bRow + k0);
    v16h b1 = load_frag(bRow + (size_t)16 * KD + k0);
    v16h b2 = load_frag(bRow + (size_t)32 * KD + k0);
    v16h b3 = load_frag(bRow + (size_t)48 * KD + k0);
    acc[0] = __builtin_amdgcn_wmma_f32_16x16x32_f16(false, a, false, b0,
                                                    (short)0, acc[0], false, false);
    acc[1] = __builtin_amdgcn_wmma_f32_16x16x32_f16(false, a, false, b1,
                                                    (short)0, acc[1], false, false);
    acc[2] = __builtin_amdgcn_wmma_f32_16x16x32_f16(false, a, false, b2,
                                                    (short)0, acc[2], false, false);
    acc[3] = __builtin_amdgcn_wmma_f32_16x16x32_f16(false, a, false, b3,
                                                    (short)0, acc[3], false, false);
  }

  // D layout (ISA 7.12.2): lane 0-15 -> M rows 0..7 in vgprs 0..7 (N = lane),
  //                        lane 16-31 -> M rows 8..15        (N = lane-16).
  const int mrow = m0 + ((lane >> 4) << 3);
  const int ncol = n0 + (lane & 15);
#pragma unroll
  for (int t = 0; t < 4; ++t) {
    const int n  = ncol + t * 16;
    const float bb = bias ? bias[n] : 0.0f;
#pragma unroll
    for (int r = 0; r < 8; ++r) {
      const int m = mrow + r;
      float v = acc[t][r] + bb;
      if (EPI == 0) {
        ((float*)outp)[(size_t)m * N + n] = v;
      } else if (EPI == 1) {
        float g = 0.5f * v * (1.0f + erff(v * 0.70710678118654752440f));
        ((_Float16*)outp)[(size_t)m * N + n] = (_Float16)g;
      } else {
        ((float*)outp)[(size_t)m * N + n] = v + res[(size_t)m * N + n];
      }
    }
  }
}

// ---------------- prep kernels ----------------
__global__ void cvt_f32_to_f16(const float* __restrict__ src,
                               _Float16* __restrict__ dst, int n) {
  int i = blockIdx.x * blockDim.x + threadIdx.x;
  if (i < n) dst[i] = (_Float16)src[i];
}

// xs = x * (gt ? 1.2 : 1.0), cast to f16. gt flat index == pixel index.
__global__ void make_xs(const float* __restrict__ x, const int* __restrict__ gt,
                        _Float16* __restrict__ dst, int n) {
  int i = blockIdx.x * blockDim.x + threadIdx.x;
  if (i < n) {
    int p = i >> 7;  // / DIM
    float s = gt[p] ? 1.2f : 1.0f;
    dst[i] = (_Float16)(x[i] * s);
  }
}

// ---------------- dilated neighborhood attention ----------------
// One wave32 per (pixel, head). lane = channel within head (0..31).
// qkv: [NPIX][384] f32 (q | k | v). Writes f16 [NPIX][128].
__global__ __launch_bounds__(256) void natten_kernel(
    const float* __restrict__ qkv, const float* __restrict__ rpb,
    _Float16* __restrict__ outH) {
  const int lane = threadIdx.x & 31;
  const int wid  = (blockIdx.x * blockDim.x + threadIdx.x) >> 5;
  if (wid >= NPIX * HEADS) return;
  const int head = wid & (HEADS - 1);
  const int p    = wid >> 2;                 // b*H*W + i*W + j
  const int j    = p % WWID;
  const int i    = (p / WWID) % HH;
  const int b    = p / (HH * WWID);

  // NATTEN dilated neighbor + relative-position indices (L=56, k=11, d=2)
  int ih[KK], rh[KK], iw[KK], rw[KK];
  {
    const int g = i % DIL, ii = i / DIL;
    const int Lg = (HH - g + DIL - 1) / DIL;
    int st = ii - KK / 2;
    if (st < 0) st = 0;
    if (st > Lg - KK) st = Lg - KK;
#pragma unroll
    for (int t = 0; t < KK; ++t) { ih[t] = g + DIL * (st + t); rh[t] = KK - 1 + st + t - ii; }
  }
  {
    const int g = j % DIL, jj = j / DIL;
    const int Lg = (WWID - g + DIL - 1) / DIL;
    int st = jj - KK / 2;
    if (st < 0) st = 0;
    if (st > Lg - KK) st = Lg - KK;
#pragma unroll
    for (int t = 0; t < KK; ++t) { iw[t] = g + DIL * (st + t); rw[t] = KK - 1 + st + t - jj; }
  }

  const float scale = 0.17677669529663687f;  // 32^-0.5
  const float q = qkv[(size_t)p * 384 + head * DH + lane] * scale;
  const float* rp = rpb + head * (2 * KK - 1) * (2 * KK - 1);

  float e[4];
  float mx = -3.4e38f;
  for (int n = 0; n < KK * KK; ++n) {
    const int a = n / KK, c = n % KK;
    const int np = b * (HH * WWID) + ih[a] * WWID + iw[c];
    float prod = q * qkv[(size_t)np * 384 + DIM + head * DH + lane];
    prod = wave_reduce_add(prod);                       // uniform across lanes
    prod += rp[rh[a] * (2 * KK - 1) + rw[c]];
    if ((n & 31) == lane) e[n >> 5] = prod;             // lane owns n%32==lane
    mx = fmaxf(mx, prod);
  }

  float s = 0.0f;
#pragma unroll
  for (int t = 0; t < 4; ++t) {
    if (t * 32 + lane < KK * KK) { e[t] = __expf(e[t] - mx); s += e[t]; }
  }
  s = wave_reduce_add(s);
  const float inv = 1.0f / s;

  float o = 0.0f;
  for (int n = 0; n < KK * KK; ++n) {
    const float w = __shfl(e[n >> 5], n & 31, 32) * inv;
    const int a = n / KK, c = n % KK;
    const int np = b * (HH * WWID) + ih[a] * WWID + iw[c];
    o += w * qkv[(size_t)np * 384 + 2 * DIM + head * DH + lane];
  }
  outH[(size_t)p * DIM + head * DH + lane] = (_Float16)o;
}

// ---------------- LayerNorm (row of 128) -> f16 ----------------
__global__ __launch_bounds__(256) void ln_kernel(
    const float* __restrict__ x1, const float* __restrict__ gam,
    const float* __restrict__ bet, _Float16* __restrict__ out) {
  const int lane = threadIdx.x & 31;
  const int wid  = (blockIdx.x * blockDim.x + threadIdx.x) >> 5;
  if (wid >= NPIX) return;
  const float* r = x1 + (size_t)wid * DIM;
  float v[4], s = 0.0f;
#pragma unroll
  for (int t = 0; t < 4; ++t) { v[t] = r[t * 32 + lane]; s += v[t]; }
  const float mean = wave_reduce_add(s) * (1.0f / DIM);
  float var = 0.0f;
#pragma unroll
  for (int t = 0; t < 4; ++t) { float d = v[t] - mean; var += d * d; }
  var = wave_reduce_add(var) * (1.0f / DIM);
  const float rs = rsqrtf(var + 1e-6f);
#pragma unroll
  for (int t = 0; t < 4; ++t) {
    const int c = t * 32 + lane;
    out[(size_t)wid * DIM + c] = (_Float16)((v[t] - mean) * rs * gam[c] + bet[c]);
  }
}

// ---------------------------------------------------------------------------
extern "C" void kernel_launch(void* const* d_in, const int* in_sizes, int n_in,
                              void* d_out, int out_size, void* d_ws, size_t ws_size,
                              hipStream_t stream) {
  (void)in_sizes; (void)n_in; (void)out_size; (void)ws_size;
  const float* x      = (const float*)d_in[0];
  const int*   gt     = (const int*)  d_in[1];
  const float* qkv_w  = (const float*)d_in[2];
  const float* rpb    = (const float*)d_in[3];
  const float* proj_w = (const float*)d_in[4];
  const float* proj_b = (const float*)d_in[5];
  const float* n2g    = (const float*)d_in[6];
  const float* n2b    = (const float*)d_in[7];
  const float* fc1_w  = (const float*)d_in[8];
  const float* fc1_b  = (const float*)d_in[9];
  const float* fc2_w  = (const float*)d_in[10];
  const float* fc2_b  = (const float*)d_in[11];
  float* out = (float*)d_out;

  // workspace carve-out (256B aligned)
  char* ws = (char*)d_ws;
  size_t off = 0;
  auto carve = [&](size_t bytes) -> void* {
    void* p = ws + off;
    off += (bytes + 255) & ~(size_t)255;
    return p;
  };
  _Float16* xs_h    = (_Float16*)carve((size_t)NPIX * DIM * 2);
  _Float16* qkvw_h  = (_Float16*)carve((size_t)3 * DIM * DIM * 2);
  _Float16* projw_h = (_Float16*)carve((size_t)DIM * DIM * 2);
  _Float16* fc1w_h  = (_Float16*)carve((size_t)MLP * DIM * 2);
  _Float16* fc2w_h  = (_Float16*)carve((size_t)DIM * MLP * 2);
  float*    qkv_f   = (float*)   carve((size_t)NPIX * 3 * DIM * 4);
  _Float16* attn_h  = (_Float16*)carve((size_t)NPIX * DIM * 2);
  float*    x1_f    = (float*)   carve((size_t)NPIX * DIM * 4);
  _Float16* ln_h    = (_Float16*)carve((size_t)NPIX * DIM * 2);
  _Float16* hid_h   = (_Float16*)carve((size_t)NPIX * MLP * 2);

  const int TPB = 256;
  auto blocks = [](int n, int t) { return (n + t - 1) / t; };

  // prep: weight f32->f16, saliency-scaled activations
  cvt_f32_to_f16<<<blocks(3 * DIM * DIM, TPB), TPB, 0, stream>>>(qkv_w, qkvw_h, 3 * DIM * DIM);
  cvt_f32_to_f16<<<blocks(DIM * DIM, TPB), TPB, 0, stream>>>(proj_w, projw_h, DIM * DIM);
  cvt_f32_to_f16<<<blocks(MLP * DIM, TPB), TPB, 0, stream>>>(fc1_w, fc1w_h, MLP * DIM);
  cvt_f32_to_f16<<<blocks(DIM * MLP, TPB), TPB, 0, stream>>>(fc2_w, fc2w_h, DIM * MLP);
  make_xs<<<blocks(NPIX * DIM, TPB), TPB, 0, stream>>>(x, gt, xs_h, NPIX * DIM);

  // GEMM grid: x covers N/64, y covers M/128 (= 49, exact)
  dim3 gQKV(3 * DIM / 64, NPIX / 128);
  wmma_gemm<0, DIM><<<gQKV, TPB, 0, stream>>>(xs_h, qkvw_h, nullptr, nullptr,
                                              (void*)qkv_f, NPIX, 3 * DIM);

  // attention: one wave per (pixel, head) -> NPIX*HEADS waves, 8 waves/block
  natten_kernel<<<(NPIX * HEADS) / 8, TPB, 0, stream>>>(qkv_f, rpb, attn_h);

  // proj + residual(x) -> x1
  dim3 gPROJ(DIM / 64, NPIX / 128);
  wmma_gemm<2, DIM><<<gPROJ, TPB, 0, stream>>>(attn_h, projw_h, proj_b, x,
                                               (void*)x1_f, NPIX, DIM);

  // LayerNorm -> f16
  ln_kernel<<<NPIX / 8, TPB, 0, stream>>>(x1_f, n2g, n2b, ln_h);

  // fc1 + gelu -> f16 hidden
  dim3 gFC1(MLP / 64, NPIX / 128);
  wmma_gemm<1, DIM><<<gFC1, TPB, 0, stream>>>(ln_h, fc1w_h, fc1_b, nullptr,
                                              (void*)hid_h, NPIX, MLP);

  // fc2 + residual(x1) -> out (f32)
  dim3 gFC2(DIM / 64, NPIX / 128);
  wmma_gemm<2, MLP><<<gFC2, TPB, 0, stream>>>(hid_h, fc2w_h, fc2_b, x1_f,
                                              (void*)out, NPIX, DIM);
}
